// Transformer_31825707663529
// MI455X (gfx1250) — matrix operand order repaired and checked by reading, not compiled, not run on hardware
//
#include <hip/hip_runtime.h>

// ---------------------------------------------------------------------------
// Transformer encoder-decoder forward for MI455X (gfx1250, wave32, WMMA).
// Dense GEMMs: double-buffered LDS bf16 tiles, software-pipelined global
// b128 staging, v_wmma_f32_16x16x32_bf16 with f32 accumulation.
// ---------------------------------------------------------------------------

#define Dm   512
#define Hf   2048
#define NHc  8
#define NLc  6
#define Bc   256
#define SXc  15
#define SYc  17
#define DHc  64
#define VYc  32000

// LDS row pitch (bf16 elements) for 32-wide K rows; 36*2B = 72B = 18 banks
// -> 16 consecutive fragment rows hit 16 distinct banks.
#define LDSP 36

typedef __attribute__((ext_vector_type(16))) __bf16 v16bf;
typedef __attribute__((ext_vector_type(2)))  __bf16 bf16x2;
typedef __attribute__((ext_vector_type(8)))  float  v8f;

// fp32 -> bf16, round-to-nearest-even
__device__ __forceinline__ __bf16 f2bf(float f) {
    unsigned u = __builtin_bit_cast(unsigned, f);
    unsigned r = u + 0x7FFFu + ((u >> 16) & 1u);
    unsigned short h = (unsigned short)(r >> 16);
    return __builtin_bit_cast(__bf16, h);
}

// A fragment (16x32 bf16) from LDS tile laid out [row][k] with pitch LDSP.
// ISA 7.12.2: m = lane&15 ; lane<16 holds K {0..7,16..23}, lane>=16 {8..15,24..31}
__device__ __forceinline__ v16bf lds_row_frag_a(const __bf16* tile, int rbase, int lane) {
    const __bf16* row = tile + (rbase + (lane & 15)) * LDSP;
    int khalf = (lane >> 4) * 8;
    v16bf f;
#pragma unroll
    for (int v = 0; v < 8; ++v) {
        int k = ((v >= 4) ? 16 : 0) + khalf + (v & 3) * 2;
        bf16x2 p = *(const bf16x2*)(row + k);
        f[2 * v]     = p[0];
        f[2 * v + 1] = p[1];
    }
    return f;
}

// B fragment (32x16 bf16, KxN) from LDS tile stored transposed [n][k], pitch LDSP.
// ISA 7.12.2: n = lane&15 ; lane<16 holds K {0..7,16..23}, lane>=16 {8..15,24..31}
__device__ __forceinline__ v16bf lds_row_frag_b(const __bf16* tile, int nbase, int lane) {
    const __bf16* row = tile + (nbase + (lane & 15)) * LDSP;
    int ko = (lane >> 4) * 8;
    v16bf f;
#pragma unroll
    for (int v = 0; v < 8; ++v) {
        bf16x2 p = *(const bf16x2*)(row + ((v >= 4) ? 16 : 0) + ko + (v & 3) * 2);
        f[2 * v]     = p[0];
        f[2 * v + 1] = p[1];
    }
    return f;
}

// C = A[MxK] * W[KxN] + bias[N], optional ReLU.
// Block: 256 threads = 8 waves; block tile 128(M) x 64(N); K-step 32.
// Wave (wm in 0..3, wn in 0..1) computes a 32x32 tile = 4 WMMA accumulators.
// Double-buffered LDS; next tile's global loads issued before the WMMAs so
// memory latency hides behind the matrix pipe. M%128==0, N%64==0, K%32==0.
__global__ __launch_bounds__(256) void gemm_wmma_kernel(
    const float* __restrict__ A, const float* __restrict__ W,
    const float* __restrict__ bias, float* __restrict__ C,
    int M, int N, int K, int relu)
{
    __shared__ __bf16 Alds[2][128 * LDSP];   // [m][k] bf16
    __shared__ __bf16 Blds[2][64 * LDSP];    // [n][k] bf16 (transposed)

    int tid  = threadIdx.x;
    int lane = tid & 31;
    int wave = tid >> 5;                  // 0..7
    int wm = wave >> 1;                   // 0..3 -> 32-row M subtile
    int wn = wave & 1;                    // 0..1 -> 32-col N subtile
    int mblk = blockIdx.y * 128;
    int nblk = blockIdx.x * 64;

    // staging geometry (per thread): A = 4 float4, B = 2 float4
    int a_row[4], a_col[4];
#pragma unroll
    for (int r = 0; r < 4; ++r) {
        int e4 = r * 256 + tid;            // 0..1023 float4 slots
        a_row[r] = e4 >> 3;                // 8 float4 per 32-wide row
        a_col[r] = (e4 & 7) * 4;
    }
    int b_k[2], b_n[2];
#pragma unroll
    for (int r = 0; r < 2; ++r) {
        int e4 = r * 256 + tid;            // 0..511 float4 slots
        b_k[r] = e4 >> 4;                  // 16 float4 per 64-wide n row
        b_n[r] = (e4 & 15) * 4;
    }

    float4 ra[4], rb[2];

    // prologue: load + stage tile 0
#pragma unroll
    for (int r = 0; r < 4; ++r)
        ra[r] = *(const float4*)(A + (size_t)(mblk + a_row[r]) * K + a_col[r]);
#pragma unroll
    for (int r = 0; r < 2; ++r)
        rb[r] = *(const float4*)(W + (size_t)b_k[r] * N + nblk + b_n[r]);
#pragma unroll
    for (int r = 0; r < 4; ++r) {
        __bf16* d = Alds[0] + a_row[r] * LDSP + a_col[r];
        d[0] = f2bf(ra[r].x); d[1] = f2bf(ra[r].y);
        d[2] = f2bf(ra[r].z); d[3] = f2bf(ra[r].w);
    }
#pragma unroll
    for (int r = 0; r < 2; ++r) {
        Blds[0][(b_n[r] + 0) * LDSP + b_k[r]] = f2bf(rb[r].x);
        Blds[0][(b_n[r] + 1) * LDSP + b_k[r]] = f2bf(rb[r].y);
        Blds[0][(b_n[r] + 2) * LDSP + b_k[r]] = f2bf(rb[r].z);
        Blds[0][(b_n[r] + 3) * LDSP + b_k[r]] = f2bf(rb[r].w);
    }
    __syncthreads();

    v8f acc00 = {}, acc01 = {}, acc10 = {}, acc11 = {};
    int buf = 0;

    for (int k0 = 0; k0 < K; k0 += 32) {
        int more = (k0 + 32 < K);
        // issue next tile's global loads (no wait yet)
        if (more) {
            int k1 = k0 + 32;
#pragma unroll
            for (int r = 0; r < 4; ++r)
                ra[r] = *(const float4*)(A + (size_t)(mblk + a_row[r]) * K + k1 + a_col[r]);
#pragma unroll
            for (int r = 0; r < 2; ++r)
                rb[r] = *(const float4*)(W + (size_t)(k1 + b_k[r]) * N + nblk + b_n[r]);
        }

        // matrix work on current buffer (hides the loads above)
        v16bf a0 = lds_row_frag_a(Alds[buf], wm * 32, lane);
        v16bf a1 = lds_row_frag_a(Alds[buf], wm * 32 + 16, lane);
        v16bf b0 = lds_row_frag_b(Blds[buf], wn * 32, lane);
        v16bf b1 = lds_row_frag_b(Blds[buf], wn * 32 + 16, lane);

        acc00 = __builtin_amdgcn_wmma_f32_16x16x32_bf16(
                    false, a0, false, b0, (short)0, acc00, false, false);
        acc01 = __builtin_amdgcn_wmma_f32_16x16x32_bf16(
                    false, a0, false, b1, (short)0, acc01, false, false);
        acc10 = __builtin_amdgcn_wmma_f32_16x16x32_bf16(
                    false, a1, false, b0, (short)0, acc10, false, false);
        acc11 = __builtin_amdgcn_wmma_f32_16x16x32_bf16(
                    false, a1, false, b1, (short)0, acc11, false, false);

        // stage prefetched tile into the other buffer
        if (more) {
            int nb = buf ^ 1;
#pragma unroll
            for (int r = 0; r < 4; ++r) {
                __bf16* d = Alds[nb] + a_row[r] * LDSP + a_col[r];
                d[0] = f2bf(ra[r].x); d[1] = f2bf(ra[r].y);
                d[2] = f2bf(ra[r].z); d[3] = f2bf(ra[r].w);
            }
#pragma unroll
            for (int r = 0; r < 2; ++r) {
                Blds[nb][(b_n[r] + 0) * LDSP + b_k[r]] = f2bf(rb[r].x);
                Blds[nb][(b_n[r] + 1) * LDSP + b_k[r]] = f2bf(rb[r].y);
                Blds[nb][(b_n[r] + 2) * LDSP + b_k[r]] = f2bf(rb[r].z);
                Blds[nb][(b_n[r] + 3) * LDSP + b_k[r]] = f2bf(rb[r].w);
            }
        }
        __syncthreads();
        buf ^= 1;
    }

    // C/D layout: VGPR v -> M = base + v + (lane>=16 ? 8 : 0); N = base + (lane&15)
    int lr  = (lane >> 4) ? 8 : 0;
    int nn0 = nblk + wn * 32 + (lane & 15);
    int nn1 = nn0 + 16;
    float bi0 = bias[nn0];
    float bi1 = bias[nn1];
#pragma unroll
    for (int v = 0; v < 8; ++v) {
        int row0 = mblk + wm * 32 + lr + v;
        int row1 = row0 + 16;
        float c00 = acc00[v] + bi0;
        float c01 = acc01[v] + bi1;
        float c10 = acc10[v] + bi0;
        float c11 = acc11[v] + bi1;
        if (relu) {
            c00 = fmaxf(c00, 0.f); c01 = fmaxf(c01, 0.f);
            c10 = fmaxf(c10, 0.f); c11 = fmaxf(c11, 0.f);
        }
        C[(size_t)row0 * N + nn0] = c00;
        C[(size_t)row0 * N + nn1] = c01;
        C[(size_t)row1 * N + nn0] = c10;
        C[(size_t)row1 * N + nn1] = c11;
    }
}

// out[row, d] = emb[tok[row], d] + PE(row % S, d); one block per token row.
__global__ __launch_bounds__(256) void embed_pe_kernel(
    const int* __restrict__ tok, const float* __restrict__ emb,
    float* __restrict__ out, int S)
{
    int row = blockIdx.x;
    int s   = row % S;
    int t   = tok[row];
    const float* er = emb + (size_t)t * Dm;
    float* orow = out + (size_t)row * Dm;
#pragma unroll
    for (int j = 0; j < 2; ++j) {
        int d = threadIdx.x + j * 256;
        float freq  = __expf(-__logf(10000.f) * (float)(d & ~1) / (float)Dm);
        float angle = (float)s * freq;
        float pe    = (d & 1) ? __cosf(angle) : __sinf(angle);
        orow[d] = er[d] + pe;
    }
}

// out = LayerNorm(x + a) * g + b ; one wave per 512-wide row.
__global__ __launch_bounds__(256) void resid_ln_kernel(
    const float* __restrict__ x, const float* __restrict__ a,
    const float* __restrict__ g, const float* __restrict__ bb,
    float* __restrict__ out, int M)
{
    int w = blockIdx.x * 8 + (threadIdx.x >> 5);
    if (w >= M) return;
    int lane = threadIdx.x & 31;
    const float* xr = x + (size_t)w * Dm;
    const float* ar = a + (size_t)w * Dm;
    float vals[16];
    float sum = 0.f;
#pragma unroll
    for (int i = 0; i < 16; ++i) {
        vals[i] = xr[lane + 32 * i] + ar[lane + 32 * i];
        sum += vals[i];
    }
#pragma unroll
    for (int off = 16; off; off >>= 1) sum += __shfl_xor(sum, off, 32);
    float mean = sum * (1.f / (float)Dm);
    float vsum = 0.f;
#pragma unroll
    for (int i = 0; i < 16; ++i) {
        float dlt = vals[i] - mean;
        vsum += dlt * dlt;
    }
#pragma unroll
    for (int off = 16; off; off >>= 1) vsum += __shfl_xor(vsum, off, 32);
    float inv = rsqrtf(vsum * (1.f / (float)Dm) + 1e-5f);
    float* orow = out + (size_t)w * Dm;
#pragma unroll
    for (int i = 0; i < 16; ++i) {
        int d = lane + 32 * i;
        orow[d] = g[d] * (vals[i] - mean) * inv + bb[d];
    }
}

// Scaled-dot-product attention with pad (+optional causal) mask.
// One wave per (b, h, qi): lanes index keys (sk <= 17 <= 32); softmax via
// shuffle reductions; each lane then produces 2 of the 64 head dims.
__global__ __launch_bounds__(256) void attn_kernel(
    const float* __restrict__ Q, const float* __restrict__ Kb,
    const float* __restrict__ Vb, const int* __restrict__ ktok,
    float* __restrict__ O, int sq, int sk, int causal)
{
    int w = blockIdx.x * 8 + (threadIdx.x >> 5);
    int total = Bc * NHc * sq;
    if (w >= total) return;
    int lane = threadIdx.x & 31;
    int qi = w % sq;
    int t  = w / sq;
    int h  = t % NHc;
    int b  = t / NHc;

    const float* qrow = Q + (size_t)(b * sq + qi) * Dm + h * DHc;

    float s = -1e30f;
    if (lane < sk) {
        const float* krow = Kb + (size_t)(b * sk + lane) * Dm + h * DHc;
        float dot = 0.f;
#pragma unroll
        for (int d = 0; d < DHc; ++d) dot += qrow[d] * krow[d];
        bool valid = (ktok[b * sk + lane] != 0) && (!causal || (lane <= qi));
        s = valid ? dot * 0.125f : -1e9f;   // 1/sqrt(64)
    }
    float mx = s;
#pragma unroll
    for (int off = 16; off; off >>= 1) mx = fmaxf(mx, __shfl_xor(mx, off, 32));
    float p = (lane < sk) ? __expf(s - mx) : 0.f;
    float sum = p;
#pragma unroll
    for (int off = 16; off; off >>= 1) sum += __shfl_xor(sum, off, 32);
    p /= sum;

    float acc0 = 0.f, acc1 = 0.f;
    int d0 = lane * 2;
    for (int kk = 0; kk < sk; ++kk) {
        float pk = __shfl(p, kk, 32);
        const float* vrow = Vb + (size_t)(b * sk + kk) * Dm + h * DHc;
        acc0 += pk * vrow[d0];
        acc1 += pk * vrow[d0 + 1];
    }
    float* orow = O + (size_t)(b * sq + qi) * Dm + h * DHc;
    orow[d0]     = acc0;
    orow[d0 + 1] = acc1;
}

// ---------------------------------------------------------------------------
// Host side
// ---------------------------------------------------------------------------
static inline void gemm(hipStream_t st, const float* A, const float* W,
                        const float* bias, float* C, int M, int N, int K, int relu) {
    dim3 grid(N / 64, M / 128);
    gemm_wmma_kernel<<<grid, 256, 0, st>>>(A, W, bias, C, M, N, K, relu);
}

static inline void attn(hipStream_t st, const float* Q, const float* Kb,
                        const float* Vb, const int* ktok, float* O,
                        int sq, int sk, int causal) {
    int waves = Bc * NHc * sq;
    attn_kernel<<<(waves + 7) / 8, 256, 0, st>>>(Q, Kb, Vb, ktok, O, sq, sk, causal);
}

static inline void resid_ln(hipStream_t st, const float* x, const float* a,
                            const float* g, const float* b, float* out, int M) {
    resid_ln_kernel<<<(M + 7) / 8, 256, 0, st>>>(x, a, g, b, out, M);
}

extern "C" void kernel_launch(void* const* d_in, const int* in_sizes, int n_in,
                              void* d_out, int out_size, void* d_ws, size_t ws_size,
                              hipStream_t stream) {
    (void)in_sizes; (void)n_in; (void)out_size; (void)ws_size;

    const int*   x          = (const int*)  d_in[0];
    const int*   y          = (const int*)  d_in[1];
    const float* emb_x      = (const float*)d_in[2];
    const float* emb_y      = (const float*)d_in[3];
    const float* enc_attn_w = (const float*)d_in[4];
    const float* enc_attn_b = (const float*)d_in[5];
    const float* enc_ln     = (const float*)d_in[6];
    const float* enc_ffn_w0 = (const float*)d_in[7];
    const float* enc_ffn_b0 = (const float*)d_in[8];
    const float* enc_ffn_w1 = (const float*)d_in[9];
    const float* enc_ffn_b1 = (const float*)d_in[10];
    const float* dec_attn_w = (const float*)d_in[11];
    const float* dec_attn_b = (const float*)d_in[12];
    const float* dec_ln     = (const float*)d_in[13];
    const float* dec_ffn_w0 = (const float*)d_in[14];
    const float* dec_ffn_b0 = (const float*)d_in[15];
    const float* dec_ffn_w1 = (const float*)d_in[16];
    const float* dec_ffn_b1 = (const float*)d_in[17];
    const float* fc_w       = (const float*)d_in[18];
    const float* fc_b       = (const float*)d_in[19];

    const int Menc = Bc * SXc;   // 3840 = 30*128
    const int Mdec = Bc * SYc;   // 4352 = 34*128
    const size_t DD = (size_t)Dm * Dm;

    // workspace carve (floats)
    float* p   = (float*)d_ws;
    float* hx  = p; p += (size_t)Menc * Dm;
    float* hy  = p; p += (size_t)Mdec * Dm;
    float* tq  = p; p += (size_t)Mdec * Dm;
    float* tk  = p; p += (size_t)Mdec * Dm;
    float* tv  = p; p += (size_t)Mdec * Dm;
    float* ta  = p; p += (size_t)Mdec * Dm;
    float* to_ = p; p += (size_t)Mdec * Dm;
    float* th0 = p; p += (size_t)Mdec * Dm;
    float* th1 = p; p += (size_t)Mdec * Dm;
    float* tff = p; p += (size_t)Mdec * Hf;

    // ---------------- encoder ----------------
    embed_pe_kernel<<<Menc, 256, 0, stream>>>(x, emb_x, hx, SXc);
    for (int i = 0; i < NLc; ++i) {
        const float* Wb  = enc_attn_w + (size_t)i * 4 * DD;
        const float* bbv = enc_attn_b + (size_t)i * 4 * Dm;
        gemm(stream, hx, Wb + 0 * DD, bbv + 0 * Dm, tq, Menc, Dm, Dm, 0);
        gemm(stream, hx, Wb + 1 * DD, bbv + 1 * Dm, tk, Menc, Dm, Dm, 0);
        gemm(stream, hx, Wb + 2 * DD, bbv + 2 * Dm, tv, Menc, Dm, Dm, 0);
        attn(stream, tq, tk, tv, x, ta, SXc, SXc, 0);
        gemm(stream, ta, Wb + 3 * DD, bbv + 3 * Dm, to_, Menc, Dm, Dm, 0);
        const float* g0 = enc_ln + (size_t)((i * 2 + 0) * 2 + 0) * Dm;
        const float* b0 = enc_ln + (size_t)((i * 2 + 0) * 2 + 1) * Dm;
        resid_ln(stream, hx, to_, g0, b0, th0, Menc);
        gemm(stream, th0, enc_ffn_w0 + (size_t)i * Dm * Hf, enc_ffn_b0 + (size_t)i * Hf,
             tff, Menc, Hf, Dm, 1);
        gemm(stream, tff, enc_ffn_w1 + (size_t)i * Hf * Dm, enc_ffn_b1 + (size_t)i * Dm,
             to_, Menc, Dm, Hf, 0);
        const float* g1 = enc_ln + (size_t)((i * 2 + 1) * 2 + 0) * Dm;
        const float* b1 = enc_ln + (size_t)((i * 2 + 1) * 2 + 1) * Dm;
        resid_ln(stream, th0, to_, g1, b1, hx, Menc);
    }

    // ---------------- decoder ----------------
    embed_pe_kernel<<<Mdec, 256, 0, stream>>>(y, emb_y, hy, SYc);
    for (int i = 0; i < NLc; ++i) {
        const float* Wb  = dec_attn_w + (size_t)i * 8 * DD;
        const float* bbv = dec_attn_b + (size_t)i * 8 * Dm;
        // masked self-attention
        gemm(stream, hy, Wb + 0 * DD, bbv + 0 * Dm, tq, Mdec, Dm, Dm, 0);
        gemm(stream, hy, Wb + 1 * DD, bbv + 1 * Dm, tk, Mdec, Dm, Dm, 0);
        gemm(stream, hy, Wb + 2 * DD, bbv + 2 * Dm, tv, Mdec, Dm, Dm, 0);
        attn(stream, tq, tk, tv, y, ta, SYc, SYc, 1);
        gemm(stream, ta, Wb + 3 * DD, bbv + 3 * Dm, to_, Mdec, Dm, Dm, 0);
        const float* g0 = dec_ln + (size_t)((i * 3 + 0) * 2 + 0) * Dm;
        const float* b0 = dec_ln + (size_t)((i * 3 + 0) * 2 + 1) * Dm;
        resid_ln(stream, hy, to_, g0, b0, th0, Mdec);
        // cross-attention (K/V from encoder output hx)
        gemm(stream, th0, Wb + 4 * DD, bbv + 4 * Dm, tq, Mdec, Dm, Dm, 0);
        gemm(stream, hx,  Wb + 5 * DD, bbv + 5 * Dm, tk, Menc, Dm, Dm, 0);
        gemm(stream, hx,  Wb + 6 * DD, bbv + 6 * Dm, tv, Menc, Dm, Dm, 0);
        attn(stream, tq, tk, tv, x, ta, SYc, SXc, 0);
        gemm(stream, ta, Wb + 7 * DD, bbv + 7 * Dm, to_, Mdec, Dm, Dm, 0);
        const float* g1 = dec_ln + (size_t)((i * 3 + 1) * 2 + 0) * Dm;
        const float* b1 = dec_ln + (size_t)((i * 3 + 1) * 2 + 1) * Dm;
        resid_ln(stream, th0, to_, g1, b1, th1, Mdec);
        // FFN
        gemm(stream, th1, dec_ffn_w0 + (size_t)i * Dm * Hf, dec_ffn_b0 + (size_t)i * Hf,
             tff, Mdec, Hf, Dm, 1);
        gemm(stream, tff, dec_ffn_w1 + (size_t)i * Hf * Dm, dec_ffn_b1 + (size_t)i * Dm,
             to_, Mdec, Dm, Hf, 0);
        const float* g2 = dec_ln + (size_t)((i * 3 + 2) * 2 + 0) * Dm;
        const float* b2 = dec_ln + (size_t)((i * 3 + 2) * 2 + 1) * Dm;
        resid_ln(stream, th1, to_, g2, b2, hy, Mdec);
    }

    // final vocab projection: [4352, 512] x [512, 32000]
    gemm(stream, hy, fc_w, fc_b, (float*)d_out, Mdec, VYc, Dm, 0);
}
